// ModelWithCRFLoss_79920751444230
// MI455X (gfx1250) — matrix operand order repaired and checked by reading, not compiled, not run on hardware
//
#include <hip/hip_runtime.h>

// CRF NLL for B=64, T=512, U=256.
// Scan step rewritten as exp-space GEMM: alpha_new = m + log( exp(alpha-m) @ exp(T) ) + pot_t
//   -> per step: (16 x 256) @ (256 x 256) per workgroup via v_wmma_f32_16x16x32_f16.
// exp(T)^T is staged once in LDS, and this wave's B tiles are then hoisted into
// persistent VGPRs (128 VGPRs/lane) for the whole 511-step scan: per step only the
// A tiles (exp(alpha-max), fresh each iteration) are read from LDS.

#define B_    64
#define T_    512
#define U_    256
#define ROWS  16          // batch rows per workgroup
#define ESTR  264         // LDS row stride (halves) for exp(T)^T  (pad 8: bank spread + 16B align)
#define AXSTR 264         // LDS row stride (halves) for exp(alpha - max)
#define ALSTR 260         // LDS row stride (floats) for alpha (1040B rows -> 16B aligned)

typedef __attribute__((ext_vector_type(16))) _Float16 v16h;
typedef __attribute__((ext_vector_type(8)))  _Float16 v8h;
typedef __attribute__((ext_vector_type(8)))  float    v8f;

__device__ __forceinline__ v16h cat8(v8h lo, v8h hi) {
  return __builtin_shufflevector(lo, hi, 0,1,2,3,4,5,6,7,8,9,10,11,12,13,14,15);
}

// ---------------------------------------------------------------------------
// Persistent forward-scan kernel: 4 blocks x 256 threads (8 waves of wave32).
// Each block owns 16 batch rows; each wave owns 2 N-tiles (32 of 256 columns).
// ---------------------------------------------------------------------------
__launch_bounds__(256, 1)
__global__ void crf_scan_kernel(const float* __restrict__ potentials,
                                const int*   __restrict__ seq_len,
                                const float* __restrict__ transitions,
                                float*       __restrict__ log_norm_out)
{
  __shared__ __align__(16) _Float16 sE [U_   * ESTR ]; // sE[j*ESTR + i]  = exp(trans[i][j])
  __shared__ __align__(16) _Float16 sAx[ROWS * AXSTR]; // exp(alpha - rowmax), f16
  __shared__ __align__(16) float    sAl[ROWS * ALSTR]; // alpha, f32
  __shared__ float sRowMax[ROWS];
  __shared__ int   sLen[ROWS];

  const int tid  = threadIdx.x;
  const int wave = tid >> 5;
  const int lane = tid & 31;
  const int grp  = lane >> 4;   // which 16-lane half
  const int ln   = lane & 15;
  const int b0   = blockIdx.x * ROWS;

  // Stage exp(transitions) transposed into LDS (coalesced global read).
  for (int e = tid; e < U_ * U_; e += 256) {
    const int i = e >> 8, j = e & 255;
    sE[j * ESTR + i] = (_Float16)__expf(transitions[e]);
  }
  // alpha0 = potentials[:, 0, :]
  for (int e = tid; e < ROWS * U_; e += 256) {
    const int m = e >> 8, j = e & 255;
    sAl[m * ALSTR + j] = potentials[(size_t)(b0 + m) * T_ * U_ + j];
  }
  if (tid < ROWS) sLen[tid] = seq_len[b0 + tid];
  __syncthreads();

  const int rowP1 = 2 * wave + grp;          // row handled in max/exp phase
  const int nt0 = wave * 2, nt1 = nt0 + 1;   // this wave's N-tiles
  const int j0 = nt0 * 16 + ln, j1 = nt1 * 16 + ln;

  // ---- hoist this wave's B tiles (constant over the whole scan) into VGPRs ----
  // B tile 32x16 f16, lane = col ln of tile: K halves [kb+grp*16 .. +15] contiguous.
  v16h bReg0[8], bReg1[8];
  #pragma unroll
  for (int kt = 0; kt < 8; ++kt) {
    const int kb = kt * 32 + grp * 16;
    bReg0[kt] = cat8(*(const v8h*)&sE[j0 * ESTR + kb],
                     *(const v8h*)&sE[j0 * ESTR + kb + 8]);
    bReg1[kt] = cat8(*(const v8h*)&sE[j1 * ESTR + kb],
                     *(const v8h*)&sE[j1 * ESTR + kb + 8]);
  }

  for (int t = 1; t < T_; ++t) {
    // ---- phase 1: per-row max, then exp(alpha - max) -> f16 A operand ----
    float v[16];
    float mx = -3.4e38f;
    #pragma unroll
    for (int q = 0; q < 4; ++q) {
      const float4 f = *(const float4*)&sAl[rowP1 * ALSTR + ln * 16 + q * 4];
      v[q*4+0] = f.x; v[q*4+1] = f.y; v[q*4+2] = f.z; v[q*4+3] = f.w;
    }
    #pragma unroll
    for (int c = 0; c < 16; ++c) mx = fmaxf(mx, v[c]);
    #pragma unroll
    for (int m = 1; m < 16; m <<= 1)
      mx = fmaxf(mx, __shfl_xor(mx, m, 32));
    if (ln == 0) sRowMax[rowP1] = mx;
    #pragma unroll
    for (int c = 0; c < 16; ++c)
      sAx[rowP1 * AXSTR + ln * 16 + c] = (_Float16)__expf(v[c] - mx);
    __syncthreads();

    // ---- phase 2: C = exp(alpha-m) @ exp(T)  via WMMA, K = 256 in 8 steps ----
    // A tile 16x32 f16: lane row = ln; K halves [kb+grp*16 .. +15] contiguous
    v16h aT[8];
    #pragma unroll
    for (int kt = 0; kt < 8; ++kt) {
      const int kb = kt * 32;
      aT[kt] = cat8(*(const v8h*)&sAx[ln * AXSTR + kb + grp * 16],
                    *(const v8h*)&sAx[ln * AXSTR + kb + grp * 16 + 8]);
    }
    v8f acc0 = {}; v8f acc1 = {};
    #pragma unroll
    for (int kt = 0; kt < 8; ++kt) {
      acc0 = __builtin_amdgcn_wmma_f32_16x16x32_f16(false, aT[kt], false, bReg0[kt],
                                                    (short)0, acc0, false, false);
      acc1 = __builtin_amdgcn_wmma_f32_16x16x32_f16(false, aT[kt], false, bReg1[kt],
                                                    (short)0, acc1, false, false);
    }

    // hint the next time-step's potentials slab toward the caches
    if (t + 1 < T_)
      __builtin_prefetch(&potentials[((size_t)(b0 + grp * 8) * T_ + (t + 1)) * U_ + j0], 0, 1);

    // ---- phase 3: alpha = frozen ? alpha : rowmax + log(C) + pot[t] ----
    #pragma unroll
    for (int rr = 0; rr < 8; ++rr) {
      const int m = rr + grp * 8;             // C/D layout: VGPR rr <-> row rr + grp*8
      const float rm = sRowMax[m];
      const int frozen = (t >= sLen[m]);
      const size_t base = ((size_t)(b0 + m) * T_ + t) * U_;
      const float n0 = rm + __logf(acc0[rr]) + potentials[base + j0];
      const float n1 = rm + __logf(acc1[rr]) + potentials[base + j1];
      const float o0 = sAl[m * ALSTR + j0];
      const float o1 = sAl[m * ALSTR + j1];
      sAl[m * ALSTR + j0] = frozen ? o0 : n0;
      sAl[m * ALSTR + j1] = frozen ? o1 : n1;
    }
    __syncthreads();
  }

  // ---- final: log_norm[b] = logsumexp_j alpha[b, j] ----
  {
    float v[16];
    float mx = -3.4e38f;
    #pragma unroll
    for (int q = 0; q < 4; ++q) {
      const float4 f = *(const float4*)&sAl[rowP1 * ALSTR + ln * 16 + q * 4];
      v[q*4+0] = f.x; v[q*4+1] = f.y; v[q*4+2] = f.z; v[q*4+3] = f.w;
    }
    #pragma unroll
    for (int c = 0; c < 16; ++c) mx = fmaxf(mx, v[c]);
    #pragma unroll
    for (int m = 1; m < 16; m <<= 1)
      mx = fmaxf(mx, __shfl_xor(mx, m, 32));
    float s = 0.f;
    #pragma unroll
    for (int c = 0; c < 16; ++c) s += __expf(v[c] - mx);
    #pragma unroll
    for (int m = 1; m < 16; m <<= 1) s += __shfl_xor(s, m, 32);
    if (ln == 0) log_norm_out[b0 + rowP1] = mx + __logf(s);
  }
}

// ---------------------------------------------------------------------------
// Path score: unary + binary, one block per batch row.
// ---------------------------------------------------------------------------
__launch_bounds__(256)
__global__ void crf_score_kernel(const float* __restrict__ potentials,
                                 const int*   __restrict__ tags,
                                 const int*   __restrict__ seq_len,
                                 const float* __restrict__ transitions,
                                 float*       __restrict__ score_out)
{
  __shared__ float red[8];
  const int b = blockIdx.x, tid = threadIdx.x;
  const int L = seq_len[b];
  float s = 0.f;
  for (int t = tid; t < T_; t += 256) {
    if (t < L) {
      const int tg = tags[b * T_ + t];
      s += potentials[((size_t)b * T_ + t) * U_ + tg];
      if (t >= 1) s += transitions[tags[b * T_ + t - 1] * U_ + tg];
    }
  }
  #pragma unroll
  for (int m = 1; m < 32; m <<= 1) s += __shfl_xor(s, m, 32);
  if ((tid & 31) == 0) red[tid >> 5] = s;
  __syncthreads();
  if (tid == 0) {
    float tot = 0.f;
    #pragma unroll
    for (int w = 0; w < 8; ++w) tot += red[w];
    score_out[b] = tot;
  }
}

// ---------------------------------------------------------------------------
// out = -mean(score - log_norm)
// ---------------------------------------------------------------------------
__global__ void crf_combine_kernel(const float* __restrict__ score,
                                   const float* __restrict__ log_norm,
                                   float*       __restrict__ out)
{
  __shared__ float red[B_];
  const int tid = threadIdx.x;
  red[tid] = score[tid] - log_norm[tid];
  __syncthreads();
  if (tid == 0) {
    float s = 0.f;
    #pragma unroll
    for (int i = 0; i < B_; ++i) s += red[i];
    out[0] = -s / (float)B_;
  }
}

extern "C" void kernel_launch(void* const* d_in, const int* in_sizes, int n_in,
                              void* d_out, int out_size, void* d_ws, size_t ws_size,
                              hipStream_t stream) {
  const float* potentials  = (const float*)d_in[0]; // (B,T,U) f32
  const int*   tags        = (const int*)  d_in[1]; // (B,T)
  const int*   seq_len     = (const int*)  d_in[2]; // (B,)
  const float* transitions = (const float*)d_in[3]; // (U,U) f32

  float* score    = (float*)d_ws;   // B floats
  float* log_norm = score + B_;     // B floats

  crf_score_kernel<<<B_, 256, 0, stream>>>(potentials, tags, seq_len, transitions, score);
  crf_scan_kernel<<<B_ / ROWS, 256, 0, stream>>>(potentials, seq_len, transitions, log_norm);
  crf_combine_kernel<<<1, B_, 0, stream>>>(score, log_norm, (float*)d_out);
}
